// Channel_SSA_Linear_9388798509224
// MI455X (gfx1250) — compile-verified, matrix operand order checked by reading, not run on hardware
//
#include <hip/hip_runtime.h>

// ---------------- problem constants ----------------
#define T_ 4
#define B_ 16
#define N_ 1024
#define C_ 512
#define H_ 8
#define D_ 64
#define M_ (T_ * B_ * N_)        // 65536 GEMM rows
#define BNC (B_ * N_ * C_)       // 8388608
#define TBNC (T_ * B_ * N_ * C_) // 33554432

#define LDS_KPAD 520             // 512 halves + 8 halves (16B) TDM row padding

typedef _Float16 v8h  __attribute__((ext_vector_type(8)));
typedef _Float16 v16h __attribute__((ext_vector_type(16)));
typedef float    v8f  __attribute__((ext_vector_type(8)));
typedef unsigned int u32x4 __attribute__((ext_vector_type(4)));
typedef int          i32x4 __attribute__((ext_vector_type(4)));
typedef int          i32x8 __attribute__((ext_vector_type(8)));

#if defined(__has_builtin)
#if __has_builtin(__builtin_amdgcn_tensor_load_to_lds)
#define HAVE_TDM 1
#endif
#endif

// ---------------------------------------------------------------------------
// Weight prep: fold BN affine into W (and bias), transpose to [n][k], split
// into f16 hi + f16 lo so spike (0/1) GEMMs recover ~fp32 accuracy.
// ---------------------------------------------------------------------------
__global__ void prep_weights(const float* __restrict__ W,
                             const float* __restrict__ bias,
                             const float* __restrict__ bnp,
                             _Float16* __restrict__ Whi_t,
                             _Float16* __restrict__ Wlo_t,
                             float* __restrict__ bias_eff,
                             int has_bias) {
  int idx = blockIdx.x * blockDim.x + threadIdx.x;   // enumerates (d, c)
  if (idx >= C_ * C_) return;
  int d = idx / C_;
  int c = idx % C_;
  float gamma = bnp[0 * C_ + d];
  float beta  = bnp[1 * C_ + d];
  float mean  = bnp[2 * C_ + d];
  float var   = bnp[3 * C_ + d];
  float s = gamma * rsqrtf(var + 1e-5f);
  float w = W[c * C_ + d] * s;
  _Float16 hi = (_Float16)w;
  Whi_t[idx] = hi;
  Wlo_t[idx] = (_Float16)(w - (float)hi);
  if (c == 0) {
    float b0 = has_bias ? bias[d] : 0.0f;
    bias_eff[d] = beta + s * (b0 - mean);
  }
}

// ---------------------------------------------------------------------------
// LIF over T for the shortcut path: x (T,B,N,C) fp32 -> xs spikes f16.
// ---------------------------------------------------------------------------
__global__ void lif_x_kernel(const float* __restrict__ x,
                             _Float16* __restrict__ xs) {
  size_t i = (size_t)blockIdx.x * blockDim.x + threadIdx.x;
  if (i >= (size_t)BNC) return;
  float v = 0.0f;
#pragma unroll
  for (int t = 0; t < T_; ++t) {
    float xv = x[(size_t)t * BNC + i];
    v += (xv - v) * 0.5f;
    float s = (v - 1.0f) >= 0.0f ? 1.0f : 0.0f;
    xs[(size_t)t * BNC + i] = (_Float16)s;
    v *= (1.0f - s);
  }
}

// ---------------------------------------------------------------------------
// LIF over T for a projection: Y (GEMM out, BN folded) -> spikes f16.
// Optionally scatters spikes as fp32 into (T,B,H,N,D) layout (the "v" output).
// ---------------------------------------------------------------------------
__global__ void lif_proj_kernel(const float* __restrict__ Y,
                                _Float16* __restrict__ spk,
                                float* __restrict__ vout, int writeV) {
  size_t i = (size_t)blockIdx.x * blockDim.x + threadIdx.x;
  if (i >= (size_t)BNC) return;
  int c = (int)(i % C_);
  size_t bn = i / C_;
  int n = (int)(bn % N_);
  int b = (int)(bn / N_);
  int h = c >> 6;
  int d = c & (D_ - 1);
  float v = 0.0f;
#pragma unroll
  for (int t = 0; t < T_; ++t) {
    float yv = Y[(size_t)t * BNC + i];
    v += (yv - v) * 0.5f;
    float s = (v - 1.0f) >= 0.0f ? 1.0f : 0.0f;
    spk[(size_t)t * BNC + i] = (_Float16)s;
    if (writeV) {
      size_t vo = ((((size_t)t * B_ + b) * H_ + h) * N_ + n) * D_ + d;
      vout[vo] = s;
    }
    v *= (1.0f - s);
  }
}

// ---------------------------------------------------------------------------
// kv = sum_n k*v per (t,b,h,d), then LIF scan (v_th = 0.5) -> kv spikes f16.
// ---------------------------------------------------------------------------
__global__ void kv_kernel(const _Float16* __restrict__ kspk,
                          const _Float16* __restrict__ vspk,
                          _Float16* __restrict__ kvspk) {
  int bh = blockIdx.x;
  int b = bh >> 3;
  int h = bh & 7;
  int d = threadIdx.x & (D_ - 1);
  int nc = threadIdx.x >> 6;
  __shared__ float red[256];
  __shared__ float kvt[T_][D_];
  for (int t = 0; t < T_; ++t) {
    float acc = 0.0f;
    size_t base = ((size_t)(t * B_ + b)) * N_ * C_ + (size_t)h * D_ + d;
    for (int n = nc; n < N_; n += 4) {
      size_t off = base + (size_t)n * C_;
      acc += (float)kspk[off] * (float)vspk[off];
    }
    red[threadIdx.x] = acc;
    __syncthreads();
    if (nc == 0)
      kvt[t][d] = red[d] + red[64 + d] + red[128 + d] + red[192 + d];
    __syncthreads();
  }
  if (threadIdx.x < D_) {
    float v = 0.0f;
#pragma unroll
    for (int t = 0; t < T_; ++t) {
      float kv = kvt[t][d];
      v += (kv - v) * 0.5f;
      float s = (v - 0.5f) >= 0.0f ? 1.0f : 0.0f;
      kvspk[(((size_t)t * B_ + b) * H_ + h) * D_ + d] = (_Float16)s;
      v *= (1.0f - s);
    }
  }
}

// ---------------------------------------------------------------------------
// att[t,b,n,c] = q_spk[t,b,n,c] * kv_spk[t,b,h,d]
// ---------------------------------------------------------------------------
__global__ void att_kernel(const _Float16* __restrict__ qspk,
                           const _Float16* __restrict__ kvspk,
                           _Float16* __restrict__ att) {
  size_t i = (size_t)blockIdx.x * blockDim.x + threadIdx.x;
  if (i >= (size_t)TBNC) return;
  int c = (int)(i & (C_ - 1));
  size_t tb = i / ((size_t)N_ * C_);
  int h = c >> 6;
  int d = c & (D_ - 1);
  float kv = (float)kvspk[(tb * H_ + h) * D_ + d];
  att[i] = (_Float16)((float)qspk[i] * kv);
}

// ---------------------------------------------------------------------------
// TDM helpers
// ---------------------------------------------------------------------------
__device__ inline unsigned lds_off(const void* p) {
  // generic -> LDS address space -> 32-bit LDS byte offset
  return (unsigned)(unsigned long long)(const __attribute__((address_space(3))) char*)p;
}

// DMA a 64-row x 512-half f16 tile (row-major, row stride 512 halves) into LDS
// with 16B padding appended per 1024B row (pad_interval=256 DW, pad_amount=4 DW)
// => padded LDS row stride = LDS_KPAD halves.
__device__ inline void tdm_load_tile64x512(unsigned lds_byte_off,
                                           const _Float16* gsrc) {
#ifdef HAVE_TDM
  unsigned long long ga = (unsigned long long)(uintptr_t)gsrc;
  u32x4 g0;
  g0.x = 1u;                                     // count=1 valid descriptor
  g0.y = lds_byte_off;                           // lds_addr
  g0.z = (unsigned)ga;                           // global_addr[31:0]
  g0.w = (unsigned)((ga >> 32) & 0x01FFFFFFull)  // global_addr[56:32]
         | (2u << 30);                           // type=2 ("image")
  const unsigned td0 = 512, td1 = 64, tl0 = 512, tl1 = 64, str0 = 512;
  i32x8 g1;
  g1[0] = (int)((1u << 16)     // data_size = 2 bytes
              | (1u << 20)     // pad_enable
              | (7u << 22)     // pad_interval = 256 DWORDs (one row)
              | (3u << 25));   // pad_amount  = 4 DWORDs (16 B)
  g1[1] = (int)((td0 & 0xFFFFu) << 16);                                // dim0 lo
  g1[2] = (int)(((td0 >> 16) & 0xFFFFu) | ((td1 & 0xFFFFu) << 16));    // dim0 hi | dim1 lo
  g1[3] = (int)(((td1 >> 16) & 0xFFFFu) | ((tl0 & 0xFFFFu) << 16));    // dim1 hi | tile0
  g1[4] = (int)(tl1 & 0xFFFFu);                                        // tile1 | tile2=0
  g1[5] = (int)str0;                                                   // dim0_stride lo
  g1[6] = 0;
  g1[7] = 0;
  i32x4 gz = {0, 0, 0, 0};
#if __clang_major__ >= 23
  i32x8 gz8 = {0, 0, 0, 0, 0, 0, 0, 0};
  __builtin_amdgcn_tensor_load_to_lds(g0, g1, gz, gz, gz8, 0);
#else
  __builtin_amdgcn_tensor_load_to_lds(g0, g1, gz, gz, 0);
#endif
#endif
}

// ---------------------------------------------------------------------------
// WMMA GEMM: out[M,512] = A[M,512] @ (Bhi + Blo) + bias (+ identity)
// B tiles (weights) are TDM-staged into LDS once per block and served by
// ds_load_b128; A spikes stream from global (L2-resident, 64 MB f16).
// Block = 256 threads = 8 wave32, wave grid 4(M) x 2(N); wave tile 32x32.
// ---------------------------------------------------------------------------
__device__ inline v16h load_frag_g(const _Float16* p) {
  v8h lo = *(const v8h*)p;
  v8h hi = *(const v8h*)(p + 16);
  return __builtin_shufflevector(lo, hi, 0, 1, 2, 3, 4, 5, 6, 7,
                                 8, 9, 10, 11, 12, 13, 14, 15);
}

template <int ADD_ID>
__global__ __launch_bounds__(256) void wmma_gemm_512(
    const _Float16* __restrict__ A, const _Float16* __restrict__ Bhi,
    const _Float16* __restrict__ Blo, const float* __restrict__ bias,
    const float* __restrict__ identity, float* __restrict__ out) {
  const int K = 512;
  __shared__ _Float16 ldsB[2][64][LDS_KPAD];  // [hi/lo][row][k], 16B row pad

  int tid = threadIdx.x;
  int wave = tid >> 5;
  int lane = tid & 31;
  int mw = wave & 3;
  int nw = wave >> 2;
  int lrow = lane & 15;
  int lsel = lane >> 4;
  size_t Rbase = (size_t)blockIdx.x * 128 + (size_t)mw * 32;
  int Cblk = blockIdx.y * 64;
  int Cbase = Cblk + nw * 32;

#ifdef HAVE_TDM
  if (wave == 0) {  // TDM is a per-wave DMA op; one wave issues both tiles
    tdm_load_tile64x512(lds_off(&ldsB[0][0][0]), Bhi + (size_t)Cblk * K);
    tdm_load_tile64x512(lds_off(&ldsB[1][0][0]), Blo + (size_t)Cblk * K);
    __builtin_amdgcn_s_wait_tensorcnt(0);
  }
#else
  {  // cooperative fallback: 256 threads copy 2 x 64 x 512 halves
    for (int it = tid; it < 2 * 64 * 64; it += 256) {
      int b = it >> 12;            // hi/lo
      int r = (it >> 6) & 63;      // row
      int k8 = it & 63;            // which v8h in the row
      const _Float16* src = (b ? Blo : Bhi) + ((size_t)(Cblk + r)) * K + k8 * 8;
      *(v8h*)&ldsB[b][r][k8 * 8] = *(const v8h*)src;
    }
  }
#endif
  __syncthreads();

  v8f acc[2][2] = {};

  for (int k0 = 0; k0 < K; k0 += 32) {
    v16h a[2], bh2[2], bl2[2];
#pragma unroll
    for (int i = 0; i < 2; ++i)
      a[i] = load_frag_g(A + (Rbase + i * 16 + lrow) * K + k0 + lsel * 8);
#pragma unroll
    for (int j = 0; j < 2; ++j) {
      int brow = nw * 32 + j * 16 + lrow;
      const _Float16* ph = &ldsB[0][brow][k0 + lsel * 8];
      const _Float16* pl = &ldsB[1][brow][k0 + lsel * 8];
      bh2[j] = load_frag_g(ph);  // ds_load_b128 x2
      bl2[j] = load_frag_g(pl);
    }
#pragma unroll
    for (int i = 0; i < 2; ++i)
#pragma unroll
      for (int j = 0; j < 2; ++j) {
        acc[i][j] = __builtin_amdgcn_wmma_f32_16x16x32_f16(
            false, a[i], false, bh2[j], (short)0, acc[i][j], false, false);
        acc[i][j] = __builtin_amdgcn_wmma_f32_16x16x32_f16(
            false, a[i], false, bl2[j], (short)0, acc[i][j], false, false);
      }
  }

  // Epilogue. C/D layout: lanes 0-15 -> N=lane, M=r; lanes 16-31 -> M=8+r.
#pragma unroll
  for (int i = 0; i < 2; ++i)
#pragma unroll
    for (int j = 0; j < 2; ++j) {
      int col = Cbase + j * 16 + lrow;
      float bcol = bias[col];
#pragma unroll
      for (int r = 0; r < 8; ++r) {
        size_t row = Rbase + i * 16 + r + 8 * lsel;
        size_t idx = row * C_ + col;
        float v = acc[i][j][r] + bcol;
        if (ADD_ID) v += identity[idx];
        out[idx] = v;
      }
    }
}

// ---------------------------------------------------------------------------
// Host-side orchestration
// ---------------------------------------------------------------------------
extern "C" void kernel_launch(void* const* d_in, const int* in_sizes, int n_in,
                              void* d_out, int out_size, void* d_ws,
                              size_t ws_size, hipStream_t stream) {
  const float* x    = (const float*)d_in[0];
  const float* Wq   = (const float*)d_in[1];
  const float* bq   = (const float*)d_in[2];
  const float* Wk   = (const float*)d_in[3];
  const float* bk   = (const float*)d_in[4];
  const float* Wv   = (const float*)d_in[5];
  const float* Wp   = (const float*)d_in[6];
  const float* bp   = (const float*)d_in[7];
  const float* bn_q = (const float*)d_in[8];
  const float* bn_k = (const float*)d_in[9];
  const float* bn_v = (const float*)d_in[10];
  const float* bn_p = (const float*)d_in[11];

  float* out_main = (float*)d_out;                  // (T,B,N,C)
  float* out_v    = (float*)d_out + (size_t)TBNC;   // (T,B,H,N,D)

  char* ws = (char*)d_ws;
  size_t off = 0;
  _Float16* xs   = (_Float16*)(ws + off); off += (size_t)TBNC * 2; // also 'att'
  float*    Ybuf = (float*)(ws + off);    off += (size_t)TBNC * 4;
  _Float16* qspk = (_Float16*)(ws + off); off += (size_t)TBNC * 2;
  _Float16* kspk = (_Float16*)(ws + off); off += (size_t)TBNC * 2;
  _Float16* vspk = (_Float16*)(ws + off); off += (size_t)TBNC * 2;
  const size_t WSZ = (size_t)C_ * C_;
  _Float16* WqHi = (_Float16*)(ws + off); off += WSZ * 2;
  _Float16* WqLo = (_Float16*)(ws + off); off += WSZ * 2;
  _Float16* WkHi = (_Float16*)(ws + off); off += WSZ * 2;
  _Float16* WkLo = (_Float16*)(ws + off); off += WSZ * 2;
  _Float16* WvHi = (_Float16*)(ws + off); off += WSZ * 2;
  _Float16* WvLo = (_Float16*)(ws + off); off += WSZ * 2;
  _Float16* WpHi = (_Float16*)(ws + off); off += WSZ * 2;
  _Float16* WpLo = (_Float16*)(ws + off); off += WSZ * 2;
  float* bqe = (float*)(ws + off); off += C_ * 4;
  float* bke = (float*)(ws + off); off += C_ * 4;
  float* bve = (float*)(ws + off); off += C_ * 4;
  float* bpe = (float*)(ws + off); off += C_ * 4;
  _Float16* kvspk = (_Float16*)(ws + off); off += (size_t)T_ * B_ * H_ * D_ * 2;

  dim3 pgrid((C_ * C_ + 255) / 256);
  prep_weights<<<pgrid, 256, 0, stream>>>(Wq, bq, bn_q, WqHi, WqLo, bqe, 1);
  prep_weights<<<pgrid, 256, 0, stream>>>(Wk, bk, bn_k, WkHi, WkLo, bke, 1);
  prep_weights<<<pgrid, 256, 0, stream>>>(Wv, nullptr, bn_v, WvHi, WvLo, bve, 0);
  prep_weights<<<pgrid, 256, 0, stream>>>(Wp, bp, bn_p, WpHi, WpLo, bpe, 1);

  lif_x_kernel<<<(BNC + 255) / 256, 256, 0, stream>>>(x, xs);

  dim3 ggrid(M_ / 128, C_ / 64);

  wmma_gemm_512<0><<<ggrid, 256, 0, stream>>>(xs, WqHi, WqLo, bqe, nullptr, Ybuf);
  lif_proj_kernel<<<(BNC + 255) / 256, 256, 0, stream>>>(Ybuf, qspk, nullptr, 0);

  wmma_gemm_512<0><<<ggrid, 256, 0, stream>>>(xs, WkHi, WkLo, bke, nullptr, Ybuf);
  lif_proj_kernel<<<(BNC + 255) / 256, 256, 0, stream>>>(Ybuf, kspk, nullptr, 0);

  wmma_gemm_512<0><<<ggrid, 256, 0, stream>>>(xs, WvHi, WvLo, bve, nullptr, Ybuf);
  lif_proj_kernel<<<(BNC + 255) / 256, 256, 0, stream>>>(Ybuf, vspk, out_v, 1);

  kv_kernel<<<B_ * H_, 256, 0, stream>>>(kspk, vspk, kvspk);

  att_kernel<<<(TBNC + 255) / 256, 256, 0, stream>>>(qspk, kvspk, xs);

  wmma_gemm_512<1><<<ggrid, 256, 0, stream>>>(xs, WpHi, WpLo, bpe, x, out_main);
}
